// Gtransformerblock_14319420965164
// MI455X (gfx1250) — compile-verified, matrix operand order checked by reading, not run on hardware
//
#include <hip/hip_runtime.h>

// ---------------------------------------------------------------------------
// Graph-transformer block for MI455X (gfx1250, wave32, WMMA bf16).
// Sizes fixed by the reference.
// ---------------------------------------------------------------------------
constexpr int NV     = 6144;
constexpr int NFEATC = 512;
constexpr int NHIDC  = 256;
constexpr int NCLSC  = 8;
constexpr int ADIMC  = 128;
constexpr float EPSF = 1e-9f;

typedef __attribute__((ext_vector_type(16))) __bf16 v16bf;
typedef __attribute__((ext_vector_type(8)))  float  v8f;

// ---- bf16 <-> f32 helpers (bit-exact RNE, no reliance on __bf16 scalar ops)
__device__ __forceinline__ unsigned short f2bf(float f) {
  unsigned int u = __float_as_uint(f);
  u += 0x7FFFu + ((u >> 16) & 1u);
  return (unsigned short)(u >> 16);
}
__device__ __forceinline__ float bf2f(unsigned short s) {
  return __uint_as_float(((unsigned int)s) << 16);
}

// Load one 16-bit WMMA fragment (A layout, 16x32 bf16): per lane, row = lane%16,
// K chunks at (lane/16)*8 + {0..7} and +16 -> two contiguous 16B loads.
union FragU { v16bf v; uint4 q[2]; };
__device__ __forceinline__ v16bf ldfrag(const unsigned short* p) {
  FragU f;
  f.q[0] = *(const uint4*)(p);
  f.q[1] = *(const uint4*)(p + 16);
  return f.v;
}

// ---------------------------------------------------------------------------
// K1: u_i = Wa_i @ Wagg[i*128:(i+1)*128, :]   ([N,3] each), cb = bias fold.
// ---------------------------------------------------------------------------
__global__ __launch_bounds__(256) void k_prep_u(
    const float* __restrict__ Wa1, const float* __restrict__ Wa2,
    const float* __restrict__ Wa3, const float* __restrict__ Wagg,
    const float* __restrict__ ba1, const float* __restrict__ ba2,
    const float* __restrict__ ba3, const float* __restrict__ bagg,
    float* __restrict__ u, float* __restrict__ cb) {
  int idx = blockIdx.x * 256 + threadIdx.x;      // i*NV + c
  if (idx < 3 * NV) {
    int i = idx / NV, c = idx % NV;
    const float* Wa = (i == 0 ? Wa1 : (i == 1 ? Wa2 : Wa3)) + (size_t)c * ADIMC;
    float s0 = 0.f, s1 = 0.f, s2 = 0.f;
    for (int a = 0; a < ADIMC; ++a) {
      float w = Wa[a];
      const float* g = Wagg + (size_t)(i * ADIMC + a) * 3;
      s0 += w * g[0]; s1 += w * g[1]; s2 += w * g[2];
    }
    float* up = u + (size_t)idx * 3;
    up[0] = s0; up[1] = s1; up[2] = s2;
  }
  if (blockIdx.x == 0 && threadIdx.x < 3) {
    int j = threadIdx.x;
    float s = bagg[j];
    for (int a = 0; a < ADIMC; ++a) {
      s += ba1[a] * Wagg[a * 3 + j]
         + ba2[a] * Wagg[(ADIMC + a) * 3 + j]
         + ba3[a] * Wagg[(2 * ADIMC + a) * 3 + j];
    }
    cb[j] = s;
  }
}

// ---------------------------------------------------------------------------
// K2: z4[r,:] = adj0[r,:]@u0 + adj1[r,:]@u1 + adj2[r,:]@u2 + cb ; nz = softmax3
// One block per row; reads each adjacency exactly once (453 MB total).
// ---------------------------------------------------------------------------
__global__ __launch_bounds__(256) void k_z4_nz(
    const float* __restrict__ adj0, const float* __restrict__ adj1,
    const float* __restrict__ adj2, const float* __restrict__ u,
    const float* __restrict__ cb, float* __restrict__ nzf,
    float* __restrict__ out_nz) {
  int r = blockIdx.x, t = threadIdx.x;
  const float* a0 = adj0 + (size_t)r * NV;
  const float* a1 = adj1 + (size_t)r * NV;
  const float* a2 = adj2 + (size_t)r * NV;
  float p0 = 0.f, p1 = 0.f, p2 = 0.f;
  for (int c0 = t * 4; c0 < NV; c0 += 256 * 4) {
    float4 x0 = *(const float4*)(a0 + c0);
    float4 x1 = *(const float4*)(a1 + c0);
    float4 x2 = *(const float4*)(a2 + c0);
#define ACC3(k, e0, e1, e2)                                              \
    {                                                                    \
      const float* up0 = u + (size_t)(c0 + (k)) * 3;                     \
      const float* up1 = up0 + (size_t)NV * 3;                           \
      const float* up2 = up1 + (size_t)NV * 3;                           \
      p0 += (e0) * up0[0] + (e1) * up1[0] + (e2) * up2[0];               \
      p1 += (e0) * up0[1] + (e1) * up1[1] + (e2) * up2[1];               \
      p2 += (e0) * up0[2] + (e1) * up1[2] + (e2) * up2[2];               \
    }
    ACC3(0, x0.x, x1.x, x2.x)
    ACC3(1, x0.y, x1.y, x2.y)
    ACC3(2, x0.z, x1.z, x2.z)
    ACC3(3, x0.w, x1.w, x2.w)
#undef ACC3
  }
  __shared__ float s0[256], s1[256], s2[256];
  s0[t] = p0; s1[t] = p1; s2[t] = p2;
  __syncthreads();
  for (int s = 128; s > 0; s >>= 1) {
    if (t < s) { s0[t] += s0[t + s]; s1[t] += s1[t + s]; s2[t] += s2[t + s]; }
    __syncthreads();
  }
  if (t == 0) {
    float z0 = s0[0] + cb[0], z1 = s1[0] + cb[1], z2 = s2[0] + cb[2];
    float m = fmaxf(z0, fmaxf(z1, z2));
    float e0 = __expf(z0 - m), e1 = __expf(z1 - m), e2 = __expf(z2 - m);
    float inv = 1.f / (e0 + e1 + e2);
    float n0 = e0 * inv, n1 = e1 * inv, n2 = e2 * inv;
    nzf[r * 3 + 0] = n0; nzf[r * 3 + 1] = n1; nzf[r * 3 + 2] = n2;
    out_nz[r * 3 + 0] = n0; out_nz[r * 3 + 1] = n1; out_nz[r * 3 + 2] = n2;
  }
}

// ---------------------------------------------------------------------------
// K3: adj[r,c] = nz[c,0]*adj0 + nz[c,1]*adj1 + nz[c,2]*adj2  (bf16 output)
// (nz indexed by COLUMN, matching torch broadcasting in the reference)
// ---------------------------------------------------------------------------
__global__ __launch_bounds__(256) void k_build_adj(
    const float* __restrict__ adj0, const float* __restrict__ adj1,
    const float* __restrict__ adj2, const float* __restrict__ nzf,
    unsigned short* __restrict__ adjb) {
  size_t base = ((size_t)blockIdx.x * 256 + threadIdx.x) * 4;
  if (base >= (size_t)NV * NV) return;
  int c = (int)(base % NV);     // NV % 4 == 0: a float4 never crosses a row
  float4 x0 = *(const float4*)(adj0 + base);
  float4 x1 = *(const float4*)(adj1 + base);
  float4 x2 = *(const float4*)(adj2 + base);
#define BLEND(k, e0, e1, e2)                                             \
  {                                                                      \
    const float* w = nzf + (size_t)(c + (k)) * 3;                        \
    adjb[base + (k)] = f2bf(w[0] * (e0) + w[1] * (e1) + w[2] * (e2));    \
  }
  BLEND(0, x0.x, x1.x, x2.x)
  BLEND(1, x0.y, x1.y, x2.y)
  BLEND(2, x0.z, x1.z, x2.z)
  BLEND(3, x0.w, x1.w, x2.w)
#undef BLEND
}

// ---------------------------------------------------------------------------
// K4: f32 -> bf16 conversion
// ---------------------------------------------------------------------------
__global__ __launch_bounds__(256) void k_cvt(const float* __restrict__ in,
                                             unsigned short* __restrict__ out,
                                             int n) {
  int i = (blockIdx.x * 256 + threadIdx.x) * 4;
  if (i + 3 < n) {
    float4 v = *(const float4*)(in + i);
    out[i + 0] = f2bf(v.x); out[i + 1] = f2bf(v.y);
    out[i + 2] = f2bf(v.z); out[i + 3] = f2bf(v.w);
  } else {
    for (int k = i; k < n; ++k) out[k] = f2bf(in[k]);
  }
}

// ---------------------------------------------------------------------------
// K5: bf16 transpose: out[c*R + r] = in[r*C + c]
// ---------------------------------------------------------------------------
__global__ __launch_bounds__(256) void k_transpose_bf16(
    const unsigned short* __restrict__ in, unsigned short* __restrict__ out,
    int R, int C) {
  __shared__ unsigned short tile[32][33];
  int tx = threadIdx.x & 31, ty = threadIdx.x >> 5;
  int bx = blockIdx.x * 32, by = blockIdx.y * 32;
  for (int i = 0; i < 32; i += 8) {
    int r = by + ty + i, c = bx + tx;
    tile[ty + i][tx] = (r < R && c < C) ? in[(size_t)r * C + c] : (unsigned short)0;
  }
  __syncthreads();
  for (int i = 0; i < 32; i += 8) {
    int r = bx + ty + i, c = by + tx;   // out is [C][R]
    if (r < C && c < R) out[(size_t)r * R + c] = tile[tx][ty + i];
  }
}

// ---------------------------------------------------------------------------
// K6: generic tiled WMMA GEMM, C = A[M,K] @ Bt[Ncol,K]^T  (bf16 in, f32 acc)
//   - optional f32 bias[Ncol] + relu   (bias/relu path)
//   - optional elementwise bf16 mask multiply on output (adj-masked QK^T path)
// Block = 256 threads = 8 waves; each wave computes 16 rows x 64 cols
// (4 v_wmma_f32_16x16x32_bf16 accumulators). Grid = (Ncol/64, M/128).
// ---------------------------------------------------------------------------
__global__ __launch_bounds__(256) void k_gemm_bf16(
    const unsigned short* __restrict__ A, const unsigned short* __restrict__ Bt,
    const float* __restrict__ bias, const unsigned short* __restrict__ mask,
    unsigned short* __restrict__ C, int M, int Ncol, int K, int relu) {
  const int lane = threadIdx.x & 31;
  const int w    = threadIdx.x >> 5;
  const int row0 = blockIdx.y * 128 + w * 16;
  const int col0 = blockIdx.x * 64;
  const int kOff = (lane >> 4) * 8;

  const unsigned short* aPtr = A + (size_t)(row0 + (lane & 15)) * K + kOff;
  const unsigned short* bp0  = Bt + (size_t)(col0 +  0 + (lane & 15)) * K + kOff;
  const unsigned short* bp1  = Bt + (size_t)(col0 + 16 + (lane & 15)) * K + kOff;
  const unsigned short* bp2  = Bt + (size_t)(col0 + 32 + (lane & 15)) * K + kOff;
  const unsigned short* bp3  = Bt + (size_t)(col0 + 48 + (lane & 15)) * K + kOff;

  v8f acc0 = {}, acc1 = {}, acc2 = {}, acc3 = {};
  for (int kk = 0; kk < K; kk += 32) {
    // prefetch the streaming A operand a few tiles ahead
    __builtin_prefetch(aPtr + kk + 512, 0, 0);
    v16bf af = ldfrag(aPtr + kk);
    v16bf b0 = ldfrag(bp0 + kk);
    v16bf b1 = ldfrag(bp1 + kk);
    v16bf b2 = ldfrag(bp2 + kk);
    v16bf b3 = ldfrag(bp3 + kk);
    acc0 = __builtin_amdgcn_wmma_f32_16x16x32_bf16(false, af, false, b0, (short)0, acc0, false, false);
    acc1 = __builtin_amdgcn_wmma_f32_16x16x32_bf16(false, af, false, b1, (short)0, acc1, false, false);
    acc2 = __builtin_amdgcn_wmma_f32_16x16x32_bf16(false, af, false, b2, (short)0, acc2, false, false);
    acc3 = __builtin_amdgcn_wmma_f32_16x16x32_bf16(false, af, false, b3, (short)0, acc3, false, false);
  }

  // Epilogue. C/D layout: VGPR g, lanes 0-15 -> row g, lanes 16-31 -> row g+8;
  // column = lane%16 within the 16-wide tile.
  const int rowHalf = (lane >> 4) << 3;
  const int colLane = lane & 15;
#pragma unroll
  for (int t = 0; t < 4; ++t) {
    v8f acc = (t == 0) ? acc0 : (t == 1) ? acc1 : (t == 2) ? acc2 : acc3;
    int col = col0 + t * 16 + colLane;
    float bv = bias ? bias[col] : 0.f;
#pragma unroll
    for (int g = 0; g < 8; ++g) {
      int row = row0 + rowHalf + g;
      size_t idx = (size_t)row * Ncol + col;
      float v = acc[g];
      if (mask) {
        v *= bf2f(mask[idx]);           // A_tilde = adj * (Q K^T)
      } else {
        v += bv;
        if (relu) v = fmaxf(v, 0.f);
      }
      C[idx] = f2bf(v);
    }
  }
}

// ---------------------------------------------------------------------------
// K8: in-place row softmax of bf16 [NV,NV] rows, fused with gcn_norm factor
// 1/(1+EPS)  (row sums of a softmax are exactly 1 -> dinv = rsqrt(1+EPS)).
// ---------------------------------------------------------------------------
__global__ __launch_bounds__(256) void k_row_softmax(unsigned short* __restrict__ att) {
  int r = blockIdx.x, t = threadIdx.x;
  unsigned short* row = att + (size_t)r * NV;
  float v[NV / 256];
  float m = -1e30f;
#pragma unroll
  for (int j = 0; j < NV / 256; ++j) {
    v[j] = bf2f(row[j * 256 + t]);
    m = fmaxf(m, v[j]);
  }
  __shared__ float sm[256];
  sm[t] = m; __syncthreads();
  for (int s = 128; s > 0; s >>= 1) {
    if (t < s) sm[t] = fmaxf(sm[t], sm[t + s]);
    __syncthreads();
  }
  float M = sm[0];
  __syncthreads();
  float acc = 0.f;
#pragma unroll
  for (int j = 0; j < NV / 256; ++j) { v[j] = __expf(v[j] - M); acc += v[j]; }
  sm[t] = acc; __syncthreads();
  for (int s = 128; s > 0; s >>= 1) {
    if (t < s) sm[t] += sm[t + s];
    __syncthreads();
  }
  float inv = 1.f / (sm[0] * (1.f + EPSF));   // softmax norm * gcn_norm
#pragma unroll
  for (int j = 0; j < NV / 256; ++j) row[j * 256 + t] = f2bf(v[j] * inv);
}

// ---------------------------------------------------------------------------
// K10: Y = X_tilde @ W2   ([NV,256] bf16 x [256,8] f32 -> [NV,8] f32). Tiny.
// ---------------------------------------------------------------------------
__global__ __launch_bounds__(256) void k_xw2(const unsigned short* __restrict__ Xt,
                                             const float* __restrict__ W2,
                                             float* __restrict__ Y) {
  int idx = blockIdx.x * 256 + threadIdx.x;   // r*8 + j
  if (idx >= NV * NCLSC) return;
  int r = idx >> 3, j = idx & 7;
  const unsigned short* xr = Xt + (size_t)r * NHIDC;
  float s = 0.f;
  for (int k = 0; k < NHIDC; ++k) s += bf2f(xr[k]) * W2[k * NCLSC + j];
  Y[idx] = s;
}

// ---------------------------------------------------------------------------
// K11: z[r,:] = adj[r,:] @ Y + b2 ; out = softmax8(z). One block per row.
// ---------------------------------------------------------------------------
__global__ __launch_bounds__(256) void k_final(
    const unsigned short* __restrict__ adjb, const float* __restrict__ Y,
    const float* __restrict__ b2, float* __restrict__ out) {
  int r = blockIdx.x, t = threadIdx.x;
  const unsigned short* ar = adjb + (size_t)r * NV;
  float p[NCLSC] = {0.f, 0.f, 0.f, 0.f, 0.f, 0.f, 0.f, 0.f};
  for (int c = t; c < NV; c += 256) {
    float a = bf2f(ar[c]);
    const float* y = Y + (size_t)c * NCLSC;
#pragma unroll
    for (int j = 0; j < NCLSC; ++j) p[j] += a * y[j];
  }
  __shared__ float red[NCLSC][256];
#pragma unroll
  for (int j = 0; j < NCLSC; ++j) red[j][t] = p[j];
  __syncthreads();
  for (int s = 128; s > 0; s >>= 1) {
    if (t < s) {
#pragma unroll
      for (int j = 0; j < NCLSC; ++j) red[j][t] += red[j][t + s];
    }
    __syncthreads();
  }
  if (t == 0) {
    float z[NCLSC];
    float m = -1e30f;
#pragma unroll
    for (int j = 0; j < NCLSC; ++j) { z[j] = red[j][0] + b2[j]; m = fmaxf(m, z[j]); }
    float s = 0.f;
#pragma unroll
    for (int j = 0; j < NCLSC; ++j) { z[j] = __expf(z[j] - m); s += z[j]; }
    float inv = 1.f / s;
#pragma unroll
    for (int j = 0; j < NCLSC; ++j) out[(size_t)r * NCLSC + j] = z[j] * inv;
  }
}

// ---------------------------------------------------------------------------
// Host orchestration
// ---------------------------------------------------------------------------
extern "C" void kernel_launch(void* const* d_in, const int* in_sizes, int n_in,
                              void* d_out, int out_size, void* d_ws, size_t ws_size,
                              hipStream_t stream) {
  (void)in_sizes; (void)n_in; (void)out_size; (void)ws_size;
  const float* adj0 = (const float*)d_in[0];
  const float* adj1 = (const float*)d_in[1];
  const float* adj2 = (const float*)d_in[2];
  const float* feat = (const float*)d_in[3];
  const float* Wa1  = (const float*)d_in[4];
  const float* ba1  = (const float*)d_in[5];
  const float* Wa2  = (const float*)d_in[6];
  const float* ba2  = (const float*)d_in[7];
  const float* Wa3  = (const float*)d_in[8];
  const float* ba3  = (const float*)d_in[9];
  const float* Wagg = (const float*)d_in[10];
  const float* bagg = (const float*)d_in[11];
  const float* W1   = (const float*)d_in[12];
  const float* b1   = (const float*)d_in[13];
  const float* Wq   = (const float*)d_in[14];
  const float* bq   = (const float*)d_in[15];
  const float* Wk   = (const float*)d_in[16];
  const float* bk   = (const float*)d_in[17];
  const float* Wv   = (const float*)d_in[18];
  const float* bv   = (const float*)d_in[19];
  const float* W2   = (const float*)d_in[20];
  const float* b2   = (const float*)d_in[21];
  float* outp = (float*)d_out;

  // ---- workspace carve-up (all offsets 256B-aligned) ----
  const size_t NN = (size_t)NV * NV;
  char* wsb = (char*)d_ws;
  size_t off = 0;
  auto carve = [&](size_t bytes) -> char* {
    char* p = wsb + off;
    off += (bytes + 255) & ~(size_t)255;
    return p;
  };
  unsigned short* adjb  = (unsigned short*)carve(NN * 2);                     // blended adjacency, bf16
  unsigned short* attb  = (unsigned short*)carve(NN * 2);                     // A_tilde / attention, bf16
  unsigned short* featb = (unsigned short*)carve((size_t)NV * NFEATC * 2);
  unsigned short* tmpb  = (unsigned short*)carve((size_t)NFEATC * NHIDC * 2); // conv staging
  unsigned short* W1t   = (unsigned short*)carve((size_t)NFEATC * NHIDC * 2);
  unsigned short* Wqt   = (unsigned short*)carve((size_t)NHIDC * NHIDC * 2);
  unsigned short* Wkt   = (unsigned short*)carve((size_t)NHIDC * NHIDC * 2);
  unsigned short* Wvt   = (unsigned short*)carve((size_t)NHIDC * NHIDC * 2);
  unsigned short* hb    = (unsigned short*)carve((size_t)NV * NHIDC * 2);
  unsigned short* htb   = (unsigned short*)carve((size_t)NV * NHIDC * 2);
  unsigned short* xb    = (unsigned short*)carve((size_t)NV * NHIDC * 2);
  unsigned short* Qb    = (unsigned short*)carve((size_t)NV * NHIDC * 2);
  unsigned short* Kb    = (unsigned short*)carve((size_t)NV * NHIDC * 2);
  unsigned short* Vb    = (unsigned short*)carve((size_t)NV * NHIDC * 2);
  unsigned short* Vtb   = (unsigned short*)carve((size_t)NV * NHIDC * 2);
  unsigned short* Xtb   = (unsigned short*)carve((size_t)NV * NHIDC * 2);
  float* Yf  = (float*)carve((size_t)NV * NCLSC * 4);
  float* u   = (float*)carve((size_t)3 * NV * 3 * 4);
  float* nzf = (float*)carve((size_t)NV * 3 * 4);
  float* cb  = (float*)carve(64);

  const dim3 blk(256);

  // 1) fold attention weights: u_i = Wa_i @ Wagg_i  (tiny)
  k_prep_u<<<dim3((3 * NV + 255) / 256), blk, 0, stream>>>(
      Wa1, Wa2, Wa3, Wagg, ba1, ba2, ba3, bagg, u, cb);

  // 2) z4 -> nz (row-dot over the three adjacencies; also writes output tail)
  k_z4_nz<<<dim3(NV), blk, 0, stream>>>(adj0, adj1, adj2, u, cb, nzf,
                                        outp + (size_t)NV * NCLSC);

  // 3) blended adjacency in bf16
  k_build_adj<<<dim3((unsigned)(NN / (4 * 256))), blk, 0, stream>>>(
      adj0, adj1, adj2, nzf, adjb);

  // 4) conversions + weight transposes (Bt layout = [Ncol][K])
  k_cvt<<<dim3((NV * NFEATC / 4 + 255) / 256), blk, 0, stream>>>(feat, featb, NV * NFEATC);
  k_cvt<<<dim3((NFEATC * NHIDC / 4 + 255) / 256), blk, 0, stream>>>(W1, tmpb, NFEATC * NHIDC);
  k_transpose_bf16<<<dim3(NHIDC / 32, NFEATC / 32), blk, 0, stream>>>(tmpb, W1t, NFEATC, NHIDC);
  k_cvt<<<dim3((NHIDC * NHIDC / 4 + 255) / 256), blk, 0, stream>>>(Wq, tmpb, NHIDC * NHIDC);
  k_transpose_bf16<<<dim3(NHIDC / 32, NHIDC / 32), blk, 0, stream>>>(tmpb, Wqt, NHIDC, NHIDC);
  k_cvt<<<dim3((NHIDC * NHIDC / 4 + 255) / 256), blk, 0, stream>>>(Wk, tmpb, NHIDC * NHIDC);
  k_transpose_bf16<<<dim3(NHIDC / 32, NHIDC / 32), blk, 0, stream>>>(tmpb, Wkt, NHIDC, NHIDC);
  k_cvt<<<dim3((NHIDC * NHIDC / 4 + 255) / 256), blk, 0, stream>>>(Wv, tmpb, NHIDC * NHIDC);
  k_transpose_bf16<<<dim3(NHIDC / 32, NHIDC / 32), blk, 0, stream>>>(tmpb, Wvt, NHIDC, NHIDC);

  // 5) h = features @ W1            [NV,512]x[512,256]
  k_gemm_bf16<<<dim3(NHIDC / 64, NV / 128), blk, 0, stream>>>(
      featb, W1t, nullptr, nullptr, hb, NV, NHIDC, NFEATC, 0);
  k_transpose_bf16<<<dim3(NHIDC / 32, NV / 32), blk, 0, stream>>>(hb, htb, NV, NHIDC);

  // 6) x = relu(adj @ h + b1)       [NV,NV]x[NV,256]
  k_gemm_bf16<<<dim3(NHIDC / 64, NV / 128), blk, 0, stream>>>(
      adjb, htb, b1, nullptr, xb, NV, NHIDC, NV, 1);

  // 7) Q/K/V = x @ W{q,k,v} + b
  k_gemm_bf16<<<dim3(NHIDC / 64, NV / 128), blk, 0, stream>>>(
      xb, Wqt, bq, nullptr, Qb, NV, NHIDC, NHIDC, 0);
  k_gemm_bf16<<<dim3(NHIDC / 64, NV / 128), blk, 0, stream>>>(
      xb, Wkt, bk, nullptr, Kb, NV, NHIDC, NHIDC, 0);
  k_gemm_bf16<<<dim3(NHIDC / 64, NV / 128), blk, 0, stream>>>(
      xb, Wvt, bv, nullptr, Vb, NV, NHIDC, NHIDC, 0);
  k_transpose_bf16<<<dim3(NHIDC / 32, NV / 32), blk, 0, stream>>>(Vb, Vtb, NV, NHIDC);

  // 8) A_tilde = adj * (Q @ K^T)    (Bt for K^T is K itself, row-major)
  k_gemm_bf16<<<dim3(NV / 64, NV / 128), blk, 0, stream>>>(
      Qb, Kb, nullptr, adjb, attb, NV, NV, NHIDC, 0);

  // 9) attention = softmax rows, fused with gcn_norm constant 1/(1+eps)
  k_row_softmax<<<dim3(NV), blk, 0, stream>>>(attb);

  // 10) X_tilde = relu(attention @ V)
  k_gemm_bf16<<<dim3(NHIDC / 64, NV / 128), blk, 0, stream>>>(
      attb, Vtb, nullptr, nullptr, Xtb, NV, NHIDC, NV, 1);

  // 11) Y = X_tilde @ W2 ; z = adj @ Y + b2 ; out = softmax8(z)
  k_xw2<<<dim3((NV * NCLSC + 255) / 256), blk, 0, stream>>>(Xtb, W2, Yf);
  k_final<<<dim3(NV), blk, 0, stream>>>(adjb, Yf, b2, outp);
}
